// SparsePromptRouter_78245714199265
// MI455X (gfx1250) — matrix operand order repaired and checked by reading, not compiled, not run on hardware
//
#include <hip/hip_runtime.h>
#include <hip/hip_bf16.h>
#include <math.h>

#define EMBED_DIM 1024
#define BATCH     8192
#define NPROMPT   4096
#define TOPK      56

typedef __attribute__((ext_vector_type(16))) __bf16 v16bf;
typedef __attribute__((ext_vector_type(8)))  float  v8f;

__device__ __forceinline__ unsigned short f32_to_bf16_rne(float f) {
    union { float f; unsigned int u; } v; v.f = f;
    unsigned int u = v.u;
    u += 0x7FFFu + ((u >> 16) & 1u);   // round to nearest even
    return (unsigned short)(u >> 16);
}

// ---------------------------------------------------------------------------
// f32 -> bf16 conversion, 4 elements per thread (all sizes are multiples of 4)
// ---------------------------------------------------------------------------
__global__ __launch_bounds__(256)
void cvt_f32_bf16(const float4* __restrict__ in, uint2* __restrict__ out, int n4) {
    int i = blockIdx.x * blockDim.x + threadIdx.x;
    if (i >= n4) return;
    float4 f = in[i];
    uint2 o;
    o.x = (unsigned)f32_to_bf16_rne(f.x) | ((unsigned)f32_to_bf16_rne(f.y) << 16);
    o.y = (unsigned)f32_to_bf16_rne(f.z) | ((unsigned)f32_to_bf16_rne(f.w) << 16);
    out[i] = o;
}

// ---------------------------------------------------------------------------
// NT GEMM via v_wmma_f32_16x16x32_bf16:  C[M,N] = scale * A[M,K] @ B[N,K]^T
// Block: 256 threads = 8 waves (2 M-waves x 4 N-waves). Block tile 128x256.
// Each wave: 64x64 tile = 4x4 fragments of 16x16, K stepped by 32.
// Requires M%128==0, N%256==0, K%32==0 (true for all call sites).
// OUT_BF16: store bf16 (projections); else store f32*scale (logits).
// ---------------------------------------------------------------------------
template <bool OUT_BF16>
__global__ __launch_bounds__(256)
void gemm_nt_wmma(const unsigned short* __restrict__ A,
                  const unsigned short* __restrict__ B,
                  void* __restrict__ Cv,
                  int M, int N, int K, float scale) {
    const int lane  = threadIdx.x & 31;
    const int wave  = threadIdx.x >> 5;   // 0..7
    const int waveM = wave >> 2;          // 0..1
    const int waveN = wave & 3;           // 0..3
    const int m0 = blockIdx.y * 128 + waveM * 64;
    const int n0 = blockIdx.x * 256 + waveN * 64;
    const int r16   = lane & 15;          // row (A) / col (B) within 16
    const int khalf = lane >> 4;          // which K-half this lane holds

    v8f acc[4][4] = {};

    for (int k = 0; k < K; k += 32) {
        v16bf a[4], b[4];
#pragma unroll
        for (int t = 0; t < 4; ++t) {
            // A fragment: 16x32, lane holds row r16; elems 0..7 = K(k+khalf*8..),
            // elems 8..15 = K(k+16+khalf*8..)  -> two 16B loads
            const unsigned short* pa = A + (size_t)(m0 + t * 16 + r16) * K + k + khalf * 8;
            union { uint4 q[2]; v16bf v; } ua;
            ua.q[0] = *reinterpret_cast<const uint4*>(pa);
            ua.q[1] = *reinterpret_cast<const uint4*>(pa + 16);
            a[t] = ua.v;
            // B fragment: 32x16 (K x N), lane holds col r16; elems 0..7 = K(k+khalf*8..),
            // elems 8..15 = K(k+16+khalf*8..)  -> two 16B loads
            const unsigned short* pb = B + (size_t)(n0 + t * 16 + r16) * K + k + khalf * 8;
            union { uint4 q[2]; v16bf v; } ub;
            ub.q[0] = *reinterpret_cast<const uint4*>(pb);
            ub.q[1] = *reinterpret_cast<const uint4*>(pb + 16);
            b[t] = ub.v;
        }
#pragma unroll
        for (int mt = 0; mt < 4; ++mt)
#pragma unroll
            for (int nt = 0; nt < 4; ++nt)
                acc[mt][nt] = __builtin_amdgcn_wmma_f32_16x16x32_bf16(
                    false, a[mt], false, b[nt], (short)0, acc[mt][nt], false, false);
    }

    // C/D layout: VGPR v, lanes 0-15 -> M=v, lanes 16-31 -> M=v+8; N = lane&15
#pragma unroll
    for (int mt = 0; mt < 4; ++mt) {
#pragma unroll
        for (int nt = 0; nt < 4; ++nt) {
            const int rbase = m0 + mt * 16 + khalf * 8;
            const int c     = n0 + nt * 16 + r16;
#pragma unroll
            for (int v = 0; v < 8; ++v) {
                float val = acc[mt][nt][v] * scale;
                if (OUT_BF16) {
                    ((unsigned short*)Cv)[(size_t)(rbase + v) * N + c] = f32_to_bf16_rne(val);
                } else {
                    ((float*)Cv)[(size_t)(rbase + v) * N + c] = val;
                }
            }
        }
    }
}

// ---------------------------------------------------------------------------
// Per-row top-56 -> softmax -> sparse scatter (in place over the logits row).
// One 256-thread block per row; row cached in LDS. Tie-break: smaller index.
// ---------------------------------------------------------------------------
#define TPB 256
__global__ __launch_bounds__(TPB)
void topk_softmax_scatter(float* __restrict__ attn, int* __restrict__ idx_out) {
    __shared__ float vals[NPROMPT];
    __shared__ float rv[TPB];
    __shared__ int   ri[TPB];
    __shared__ float topv[TOPK];
    __shared__ int   topi[TOPK];
    __shared__ float probs[TOPK];

    const size_t rowbase = (size_t)blockIdx.x * NPROMPT;
    for (int j = threadIdx.x; j < NPROMPT; j += TPB) vals[j] = attn[rowbase + j];
    __syncthreads();

    for (int t = 0; t < TOPK; ++t) {
        float bv = -INFINITY; int bi = 0x7FFFFFFF;
        for (int j = threadIdx.x; j < NPROMPT; j += TPB) {
            float v = vals[j];
            if (v > bv || (v == bv && j < bi)) { bv = v; bi = j; }
        }
        rv[threadIdx.x] = bv; ri[threadIdx.x] = bi;
        __syncthreads();
        for (int s = TPB / 2; s > 0; s >>= 1) {
            if (threadIdx.x < s) {
                float ov = rv[threadIdx.x + s]; int oi = ri[threadIdx.x + s];
                if (ov > rv[threadIdx.x] || (ov == rv[threadIdx.x] && oi < ri[threadIdx.x])) {
                    rv[threadIdx.x] = ov; ri[threadIdx.x] = oi;
                }
            }
            __syncthreads();
        }
        if (threadIdx.x == 0) {
            topv[t] = rv[0]; topi[t] = ri[0];
            vals[ri[0]] = -INFINITY;
        }
        __syncthreads();
    }

    if (threadIdx.x == 0) {
        float mx = topv[0];           // iterative argmax -> topv[0] is the max
        float sum = 0.f;
        for (int t = 0; t < TOPK; ++t) { float e = expf(topv[t] - mx); probs[t] = e; sum += e; }
        float inv = 1.f / sum;
        for (int t = 0; t < TOPK; ++t) probs[t] *= inv;
    }
    __syncthreads();

    for (int j = threadIdx.x; j < NPROMPT; j += TPB) attn[rowbase + j] = 0.f;
    __syncthreads();

    if (threadIdx.x < TOPK) {
        attn[rowbase + topi[threadIdx.x]] = probs[threadIdx.x];
        idx_out[(size_t)blockIdx.x * TOPK + threadIdx.x] = topi[threadIdx.x];
    }
}

// ---------------------------------------------------------------------------
// Launch: convert -> Qh -> Kh -> logits (into d_out) -> topk/softmax/scatter
// ---------------------------------------------------------------------------
extern "C" void kernel_launch(void* const* d_in, const int* in_sizes, int n_in,
                              void* d_out, int out_size, void* d_ws, size_t ws_size,
                              hipStream_t stream) {
    const float* img  = (const float*)d_in[0];  // [8192,1024]
    const float* bank = (const float*)d_in[1];  // [4096,1024]
    const float* Wq   = (const float*)d_in[2];  // [1024,1024]
    const float* Wk   = (const float*)d_in[3];  // [1024,1024]

    unsigned short* ws     = (unsigned short*)d_ws;
    unsigned short* img_h  = ws;
    unsigned short* bank_h = img_h  + (size_t)BATCH   * EMBED_DIM;
    unsigned short* Wq_h   = bank_h + (size_t)NPROMPT * EMBED_DIM;
    unsigned short* Wk_h   = Wq_h   + (size_t)EMBED_DIM * EMBED_DIM;
    unsigned short* Q_h    = Wk_h   + (size_t)EMBED_DIM * EMBED_DIM;
    unsigned short* K_h    = Q_h    + (size_t)BATCH   * EMBED_DIM;

    auto cvt = [&](const float* src, unsigned short* dst, size_t n) {
        int n4 = (int)(n / 4);
        cvt_f32_bf16<<<(n4 + 255) / 256, 256, 0, stream>>>(
            (const float4*)src, (uint2*)dst, n4);
    };
    cvt(img,  img_h,  (size_t)BATCH   * EMBED_DIM);
    cvt(bank, bank_h, (size_t)NPROMPT * EMBED_DIM);
    cvt(Wq,   Wq_h,   (size_t)EMBED_DIM * EMBED_DIM);
    cvt(Wk,   Wk_h,   (size_t)EMBED_DIM * EMBED_DIM);

    dim3 blk(256);
    // Q = img @ Wq^T   [8192,1024], bf16 out
    gemm_nt_wmma<true><<<dim3(EMBED_DIM / 256, BATCH / 128), blk, 0, stream>>>(
        img_h, Wq_h, Q_h, BATCH, EMBED_DIM, EMBED_DIM, 1.0f);
    // K = bank @ Wk^T  [4096,1024], bf16 out
    gemm_nt_wmma<true><<<dim3(EMBED_DIM / 256, NPROMPT / 128), blk, 0, stream>>>(
        bank_h, Wk_h, K_h, NPROMPT, EMBED_DIM, EMBED_DIM, 1.0f);

    float* attn   = (float*)d_out;                       // [8192,4096]
    int*   idxout = (int*)(attn + (size_t)BATCH * NPROMPT); // [8192,56]

    // logits = Q @ K^T / sqrt(1024)   (1/32) -> f32 directly into d_out
    gemm_nt_wmma<false><<<dim3(NPROMPT / 256, BATCH / 128), blk, 0, stream>>>(
        Q_h, K_h, attn, BATCH, NPROMPT, EMBED_DIM, 1.0f / 32.0f);

    topk_softmax_scatter<<<BATCH, TPB, 0, stream>>>(attn, idxout);
}
